// SwinBlock_50251117363612
// MI455X (gfx1250) — matrix operand order repaired and checked by reading, not compiled, not run on hardware
//
#include <hip/hip_runtime.h>
#include <hip/hip_bf16.h>

typedef _Float16 half_t;
typedef __attribute__((ext_vector_type(16))) _Float16 v16h;
typedef __attribute__((ext_vector_type(4)))  _Float16 v4h;
typedef __attribute__((ext_vector_type(8)))  float    v8f;
typedef unsigned int v4u __attribute__((ext_vector_type(4)));
typedef int          v4i __attribute__((ext_vector_type(4)));
typedef int          v8i __attribute__((ext_vector_type(8)));

union Frag16 {
    v16h  v;
    uint4 u[2];
    half_t h[16];
};

__device__ __forceinline__ v8f wmma_f16(v16h a, v16h b, v8f c) {
    return __builtin_amdgcn_wmma_f32_16x16x32_f16(false, a, false, b, (short)0, c, false, false);
}

// ---------------------------------------------------------------------------
// TDM: async 2-D tile load (2-byte elements) global -> LDS.
//   rows      : tile_dim1 (number of rows)
//   rowlen    : tile_dim0 (elements per row, contiguous in memory)
//   rowstride : stride between rows in elements
// Tile is packed contiguously into LDS at lds_off.
// ---------------------------------------------------------------------------
__device__ __forceinline__ void tdm_load_2d_f16(const void* gaddr, unsigned lds_off,
                                                unsigned rows, unsigned rowlen,
                                                unsigned rowstride) {
    unsigned long long ga = (unsigned long long)(uintptr_t)gaddr;
    v4u g0;
    g0[0] = 1u;                                             // count=1, user mode
    g0[1] = lds_off;                                        // lds_addr
    g0[2] = (unsigned)(ga & 0xffffffffu);                   // global_addr[31:0]
    g0[3] = (unsigned)((ga >> 32) & 0x01ffffffu) | (2u << 30); // global_addr[56:32] | type=2
    v8i g1;
    g1[0] = (int)(1u << 16);                                // data_size=1 (2 bytes)
    g1[1] = (int)((rowlen & 0xffffu) << 16);                // tensor_dim0[15:0]
    g1[2] = (int)(((rowlen >> 16) & 0xffffu) | ((rows & 0xffffu) << 16)); // dim0 hi | dim1 lo
    g1[3] = (int)(((rows >> 16) & 0xffffu) | ((rowlen & 0xffffu) << 16)); // dim1 hi | tile_dim0
    g1[4] = (int)(rows & 0xffffu);                          // tile_dim1 | tile_dim2=0
    g1[5] = (int)rowstride;                                 // tensor_dim0_stride[31:0]
    g1[6] = 0;
    g1[7] = 0;
    v4i gz = {0, 0, 0, 0};
#if defined(__clang_major__) && (__clang_major__ >= 23)
    v8i gz8 = {0, 0, 0, 0, 0, 0, 0, 0};
    __builtin_amdgcn_tensor_load_to_lds(g0, g1, gz, gz, gz8, 0);
#else
    __builtin_amdgcn_tensor_load_to_lds(g0, g1, gz, gz, 0);
#endif
}

// ---------------------------------------------------------------------------
// Weight transpose + f32->f16 convert: Wt[n*K+k] = (f16) W[k*N+n]
// ---------------------------------------------------------------------------
__global__ void wtcvt_kernel(const float* __restrict__ W, half_t* __restrict__ Wt,
                             int K, int N) {
    int idx = blockIdx.x * 256 + threadIdx.x;
    if (idx >= K * N) return;
    int n = idx / K, k = idx - n * K;
    Wt[idx] = (half_t)W[(size_t)k * N + n];
}

// ---------------------------------------------------------------------------
// Dense relative-position bias, interleaved for float4 epilogue loads:
//   biasM[h*4096 + n*64 + (m&15)*4 + (m>>4)]  == bias(h, n, m)
// ---------------------------------------------------------------------------
__global__ void biaspre_kernel(const float* __restrict__ table, float* __restrict__ biasM) {
    int idx = blockIdx.x * 256 + threadIdx.x;     // 4096 = 64*64
    if (idx >= 4096) return;
    int n = idx >> 6, m = idx & 63;
    int dy = (n >> 3) - (m >> 3) + 7;
    int dx = (n & 7) - (m & 7) + 7;
    int r  = dy * 15 + dx;
    int dst = n * 64 + (m & 15) * 4 + (m >> 4);
    #pragma unroll
    for (int h = 0; h < 4; ++h)
        biasM[h * 4096 + dst] = table[r * 4 + h];
}

// ---------------------------------------------------------------------------
// LayerNorm over C=128, one wave (32 lanes x 4 elems) per token, f32 -> f16
// ---------------------------------------------------------------------------
__global__ __launch_bounds__(256) void ln_kernel(const float* __restrict__ x,
                                                 const float* __restrict__ g,
                                                 const float* __restrict__ b,
                                                 half_t* __restrict__ out, int ntok) {
    int wave = threadIdx.x >> 5, lane = threadIdx.x & 31;
    int t = blockIdx.x * 8 + wave;
    if (t >= ntok) return;
    const float4 xv = *(const float4*)(x + (size_t)t * 128 + lane * 4);
    float s  = xv.x + xv.y + xv.z + xv.w;
    float sq = xv.x * xv.x + xv.y * xv.y + xv.z * xv.z + xv.w * xv.w;
    #pragma unroll
    for (int m = 1; m < 32; m <<= 1) {
        s  += __shfl_xor(s,  m, 32);
        sq += __shfl_xor(sq, m, 32);
    }
    float mu   = s * (1.0f / 128.0f);
    float var  = sq * (1.0f / 128.0f) - mu * mu;
    float rstd = rsqrtf(var + 1e-5f);
    const float4 gv = *(const float4*)(g + lane * 4);
    const float4 bv = *(const float4*)(b + lane * 4);
    v4h o;
    o[0] = (half_t)((xv.x - mu) * rstd * gv.x + bv.x);
    o[1] = (half_t)((xv.y - mu) * rstd * gv.y + bv.y);
    o[2] = (half_t)((xv.z - mu) * rstd * gv.z + bv.z);
    o[3] = (half_t)((xv.w - mu) * rstd * gv.w + bv.w);
    *(v4h*)(out + (size_t)t * 128 + lane * 4) = o;
}

// ---------------------------------------------------------------------------
// WMMA GEMM:  C[M,N] = epi( A[M,K]_f16 * Bt[N,K]_f16^T + bias[N] )
// block = 256 threads = 8 waves; wave computes a 16x64 tile.
// B tiles (64 x 128 halves) staged into LDS via the Tensor Data Mover.
// EPI 0: store f16      EPI 1: + res (f32), store f32      EPI 2: GELU, store f16
// ---------------------------------------------------------------------------
template <int EPI>
__global__ __launch_bounds__(256) void gemm_kernel(const half_t* __restrict__ A,
                                                   const half_t* __restrict__ Bt,
                                                   const float* __restrict__ bias,
                                                   const float* __restrict__ res,
                                                   half_t* __restrict__ Oh,
                                                   float* __restrict__ Of,
                                                   int M, int N, int K) {
    __shared__ __align__(16) half_t bt[64 * 128];   // 16 KB B-tile stage
    int wave = threadIdx.x >> 5, lane = threadIdx.x & 31;
    int lh = lane >> 4, ll = lane & 15;
    int row_base = blockIdx.y * 128 + wave * 16;
    int col_base = blockIdx.x * 64;
    const half_t* arow = A + (size_t)(row_base + ll) * K;

    v8f zero = {0.f, 0.f, 0.f, 0.f, 0.f, 0.f, 0.f, 0.f};
    v8f acc[4] = {zero, zero, zero, zero};

    for (int kc = 0; kc < K; kc += 128) {
        __syncthreads();                    // previous chunk fully consumed
        if (wave == 0) {
            tdm_load_2d_f16(Bt + (size_t)col_base * K + kc,
                            (unsigned)(uintptr_t)&bt[0], 64u, 128u, (unsigned)K);
            __builtin_amdgcn_s_wait_tensorcnt(0);
        }
        __syncthreads();                    // publish B tile
        #pragma unroll
        for (int k0 = 0; k0 < 128; k0 += 32) {
            Frag16 a;
            a.u[0] = *(const uint4*)(arow + kc + k0 + 8 * lh);
            a.u[1] = *(const uint4*)(arow + kc + k0 + 8 * lh + 16);
            #pragma unroll
            for (int nt = 0; nt < 4; ++nt) {
                const half_t* brow = &bt[(nt * 16 + ll) * 128 + k0 + 8 * lh];
                Frag16 b;
                b.u[0] = *(const uint4*)(brow);
                b.u[1] = *(const uint4*)(brow + 16);
                acc[nt] = wmma_f16(a.v, b.v, acc[nt]);
            }
        }
    }

    #pragma unroll
    for (int nt = 0; nt < 4; ++nt) {
        int c = col_base + nt * 16 + ll;
        float bz = bias[c];
        #pragma unroll
        for (int i = 0; i < 8; ++i) {
            int r = row_base + i + 8 * lh;
            float v = acc[nt][i] + bz;
            if constexpr (EPI == 0) {
                Oh[(size_t)r * N + c] = (half_t)v;
            } else if constexpr (EPI == 1) {
                Of[(size_t)r * N + c] = v + res[(size_t)r * N + c];
            } else {
                float ge = 0.5f * v * (1.0f + erff(v * 0.70710678118f));
                Oh[(size_t)r * N + c] = (half_t)ge;
            }
        }
    }
}

// ---------------------------------------------------------------------------
// Window attention: one block (128 thr) per 8x8 window, one wave per head.
// qkv rows are token-indexed [131072, 384]; output written token-indexed.
// ---------------------------------------------------------------------------
__global__ __launch_bounds__(128) void attn_kernel(const half_t* __restrict__ qkv,
                                                   const float* __restrict__ biasM,
                                                   half_t* __restrict__ obuf) {
    __shared__ __align__(16) half_t pl[4][16 * 64];
    int w = blockIdx.x;                         // 0..2047
    int head = threadIdx.x >> 5, lane = threadIdx.x & 31;
    int lh = lane >> 4, ll = lane & 15;
    int b = w >> 6, wy = (w >> 3) & 7, wx = w & 7;
    int gbase = b * 4096 + wy * 512 + wx * 8;   // tok(n) = gbase + (n>>3)*64 + (n&7)
    const int QO = head * 32, KO = 128 + head * 32, VO = 256 + head * 32;

    // K as B-operand fragments (4 tiles of 32x16): two 16B chunks per lane at 8*lh and 8*lh+16
    v16h kb[4];
    #pragma unroll
    for (int nt = 0; nt < 4; ++nt) {
        int m = nt * 16 + ll;
        const half_t* kr = qkv + (size_t)(gbase + (m >> 3) * 64 + (m & 7)) * 384 + KO + 8 * lh;
        Frag16 f;
        f.u[0] = *(const uint4*)(kr);
        f.u[1] = *(const uint4*)(kr + 16);
        kb[nt] = f.v;
    }
    // V as B-operand fragments: [kchunk][dtile], element i <-> m = kc*32+8*lh+(i&7)+16*(i>>3)
    v16h vb[2][2];
    #pragma unroll
    for (int kc = 0; kc < 2; ++kc)
        #pragma unroll
        for (int dt = 0; dt < 2; ++dt) {
            Frag16 f;
            #pragma unroll
            for (int i = 0; i < 16; ++i) {
                int m = kc * 32 + 8 * lh + (i & 7) + ((i >> 3) << 4);
                f.h[i] = qkv[(size_t)(gbase + (m >> 3) * 64 + (m & 7)) * 384 + VO + dt * 16 + ll];
            }
            vb[kc][dt] = f.v;
        }

    const float scale = 0.17677669529663687f;   // 1/sqrt(32)
    half_t* P = &pl[head][0];

    for (int mt = 0; mt < 4; ++mt) {
        // Q A-fragment for this 16-row strip
        int qrow = mt * 16 + ll;
        const half_t* qr = qkv + (size_t)(gbase + (qrow >> 3) * 64 + (qrow & 7)) * 384 + QO + 8 * lh;
        Frag16 qa;
        qa.u[0] = *(const uint4*)(qr);
        qa.u[1] = *(const uint4*)(qr + 16);

        v8f zero = {0.f, 0.f, 0.f, 0.f, 0.f, 0.f, 0.f, 0.f};
        v8f acc[4] = {zero, zero, zero, zero};
        #pragma unroll
        for (int nt = 0; nt < 4; ++nt) acc[nt] = wmma_f16(qa.v, kb[nt], acc[nt]);

        // bias + scale + row softmax (row spread over 16 lanes of one half + 4 nt)
        float p[4][8];
        #pragma unroll
        for (int i = 0; i < 8; ++i) {
            int r = mt * 16 + i + 8 * lh;
            const float4 bq = *(const float4*)(biasM + head * 4096 + r * 64 + ll * 4);
            const float barr[4] = {bq.x, bq.y, bq.z, bq.w};
            float mx = -3.0e38f;
            #pragma unroll
            for (int nt = 0; nt < 4; ++nt) {
                float s = acc[nt][i] * scale + barr[nt];
                p[nt][i] = s;
                mx = fmaxf(mx, s);
            }
            #pragma unroll
            for (int msk = 1; msk < 16; msk <<= 1) mx = fmaxf(mx, __shfl_xor(mx, msk, 32));
            float sm = 0.0f;
            #pragma unroll
            for (int nt = 0; nt < 4; ++nt) {
                float e = __expf(p[nt][i] - mx);
                p[nt][i] = e;
                sm += e;
            }
            #pragma unroll
            for (int msk = 1; msk < 16; msk <<= 1) sm += __shfl_xor(sm, msk, 32);
            float inv = 1.0f / sm;
            #pragma unroll
            for (int nt = 0; nt < 4; ++nt) p[nt][i] *= inv;
        }

        // stage P in LDS to re-shape C-layout -> A-layout
        #pragma unroll
        for (int nt = 0; nt < 4; ++nt)
            #pragma unroll
            for (int i = 0; i < 8; ++i)
                P[(i + 8 * lh) * 64 + nt * 16 + ll] = (half_t)p[nt][i];
        __syncthreads();

        Frag16 pa0, pa1;
        {
            const half_t* pr = P + ll * 64 + 8 * lh;
            pa0.u[0] = *(const uint4*)(pr);
            pa0.u[1] = *(const uint4*)(pr + 16);
            pa1.u[0] = *(const uint4*)(pr + 32);
            pa1.u[1] = *(const uint4*)(pr + 48);
        }

        v8f oacc[2] = {zero, zero};
        #pragma unroll
        for (int dt = 0; dt < 2; ++dt) {
            oacc[dt] = wmma_f16(pa0.v, vb[0][dt], oacc[dt]);
            oacc[dt] = wmma_f16(pa1.v, vb[1][dt], oacc[dt]);
        }
        #pragma unroll
        for (int dt = 0; dt < 2; ++dt)
            #pragma unroll
            for (int i = 0; i < 8; ++i) {
                int n = mt * 16 + i + 8 * lh;
                obuf[(size_t)(gbase + (n >> 3) * 64 + (n & 7)) * 128 + head * 32 + dt * 16 + ll] =
                    (half_t)oacc[dt][i];
            }
        __syncthreads();
    }
}

// ---------------------------------------------------------------------------
extern "C" void kernel_launch(void* const* d_in, const int* in_sizes, int n_in,
                              void* d_out, int out_size, void* d_ws, size_t ws_size,
                              hipStream_t stream) {
    (void)in_sizes; (void)n_in; (void)out_size; (void)ws_size;
    const float* x          = (const float*)d_in[0];
    const float* W_qkv      = (const float*)d_in[1];
    const float* b_qkv      = (const float*)d_in[2];
    const float* W_out      = (const float*)d_in[3];
    const float* b_out      = (const float*)d_in[4];
    const float* bias_table = (const float*)d_in[5];
    const float* g1         = (const float*)d_in[6];
    const float* be1        = (const float*)d_in[7];
    const float* g2         = (const float*)d_in[8];
    const float* be2        = (const float*)d_in[9];
    const float* W_fc1      = (const float*)d_in[10];
    const float* b_fc1      = (const float*)d_in[11];
    const float* W_fc2      = (const float*)d_in[12];
    const float* b_fc2      = (const float*)d_in[13];
    float* out = (float*)d_out;

    const int M = 32 * 4096;   // tokens

    // workspace layout (bytes)
    char* ws = (char*)d_ws;
    const size_t OFF_QKV = 0;                                // f16 [M,384]  96 MB
    const size_t OFF_OB  = OFF_QKV + (size_t)M * 384 * 2;    // f16 [M,128]  32 MB
    const size_t OFF_XN  = OFF_OB  + (size_t)M * 128 * 2;    // f16 [M,128]  32 MB (xn1 then xn2)
    const size_t OFF_X2  = OFF_XN  + (size_t)M * 128 * 2;    // f32 [M,128]  64 MB
    const size_t OFF_WQ  = OFF_X2  + (size_t)M * 128 * 4;    // WqkvT f16 384x128
    const size_t OFF_WO  = OFF_WQ  + 384 * 128 * 2;          // WoutT f16 128x128
    const size_t OFF_W1  = OFF_WO  + 128 * 128 * 2;          // Wfc1T f16 512x128
    const size_t OFF_W2  = OFF_W1  + 512 * 128 * 2;          // Wfc2T f16 128x512
    const size_t OFF_BM  = OFF_W2  + 128 * 512 * 2;          // biasM f32 4x64x64
    const size_t OFF_H   = OFF_QKV;                          // f16 [M,512] reuses qkv+obuf (128 MB)

    half_t* qkvb  = (half_t*)(ws + OFF_QKV);
    half_t* obuf  = (half_t*)(ws + OFF_OB);
    half_t* xn    = (half_t*)(ws + OFF_XN);
    float*  x2    = (float*) (ws + OFF_X2);
    half_t* WqkvT = (half_t*)(ws + OFF_WQ);
    half_t* WoutT = (half_t*)(ws + OFF_WO);
    half_t* Wfc1T = (half_t*)(ws + OFF_W1);
    half_t* Wfc2T = (half_t*)(ws + OFF_W2);
    float*  biasM = (float*) (ws + OFF_BM);
    half_t* hbuf  = (half_t*)(ws + OFF_H);

    // 1. weight prep + dense attention bias
    wtcvt_kernel<<<(128 * 384 + 255) / 256, 256, 0, stream>>>(W_qkv, WqkvT, 128, 384);
    wtcvt_kernel<<<(128 * 128 + 255) / 256, 256, 0, stream>>>(W_out, WoutT, 128, 128);
    wtcvt_kernel<<<(128 * 512 + 255) / 256, 256, 0, stream>>>(W_fc1, Wfc1T, 128, 512);
    wtcvt_kernel<<<(512 * 128 + 255) / 256, 256, 0, stream>>>(W_fc2, Wfc2T, 512, 128);
    biaspre_kernel<<<16, 256, 0, stream>>>(bias_table, biasM);

    // 2. LN1
    ln_kernel<<<M / 8, 256, 0, stream>>>(x, g1, be1, xn, M);

    // 3. QKV GEMM [M,128]x[128,384] -> f16
    gemm_kernel<0><<<dim3(384 / 64, M / 128), 256, 0, stream>>>(
        xn, WqkvT, b_qkv, nullptr, qkvb, nullptr, M, 384, 128);

    // 4. window attention
    attn_kernel<<<2048, 128, 0, stream>>>(qkvb, biasM, obuf);

    // 5. out-proj + residual -> x2 (f32)
    gemm_kernel<1><<<dim3(128 / 64, M / 128), 256, 0, stream>>>(
        obuf, WoutT, b_out, x, nullptr, x2, M, 128, 128);

    // 6. LN2
    ln_kernel<<<M / 8, 256, 0, stream>>>(x2, g2, be2, xn, M);

    // 7. FC1 + GELU -> f16
    gemm_kernel<2><<<dim3(512 / 64, M / 128), 256, 0, stream>>>(
        xn, Wfc1T, b_fc1, nullptr, hbuf, nullptr, M, 512, 128);

    // 8. FC2 + residual -> d_out (f32)
    gemm_kernel<1><<<dim3(128 / 64, M / 128), 256, 0, stream>>>(
        hbuf, Wfc2T, b_fc2, x2, nullptr, out, M, 128, 512);
}
